// behavior_model_74569222193474
// MI455X (gfx1250) — compile-verified, matrix operand order checked by reading, not run on hardware
//
#include <hip/hip_runtime.h>
#include <hip/hip_bf16.h>

typedef unsigned int u32;
typedef unsigned short u16;

typedef __attribute__((ext_vector_type(16))) __bf16 v16bf;
typedef __attribute__((ext_vector_type(8)))  float  v8f;

union Frag { u32 u[8]; v16bf v; };

// ---------- scalar bf16 helpers (RNE) ----------
__device__ __forceinline__ u16 f2bf(float f) {
  union { float f; u32 u; } x; x.f = f;
  u32 u = x.u;
  u32 r = u + 0x7FFFu + ((u >> 16) & 1u);
  return (u16)(r >> 16);
}
__device__ __forceinline__ float bf2f(u16 v) {
  union { u32 u; float f; } x; x.u = ((u32)v) << 16;
  return x.f;
}
__device__ __forceinline__ u32 pack2(float a, float b) {
  return (u32)f2bf(a) | ((u32)f2bf(b) << 16);
}

// ---------- WMMA helpers ----------
// 16-bit operand fragment layout (CDNA5 ISA 7.12.2):
// lane L (L<16): row = L, K elements {k0..k0+7} U {k0+16..k0+23}
// lane L (L>=16): row = L-16, K base +8. Pairs of consecutive K share a dword.
__device__ __forceinline__ v16bf load_frag_rm(const u16* __restrict__ base, int ld,
                                              int r0, int k0, int lane) {
  int r  = r0 + (lane & 15);
  int kb = k0 + ((lane & 16) ? 8 : 0);
  const u32* p = (const u32*)(base + (size_t)r * ld + kb);
  Frag f;
#pragma unroll
  for (int v = 0; v < 8; ++v) f.u[v] = p[(v & 3) + ((v >> 2) << 3)];
  return f.v;
}

__device__ __forceinline__ v8f wmma_bf16(v16bf a, v16bf b, v8f c) {
  return __builtin_amdgcn_wmma_f32_16x16x32_bf16(false, a, false, b, (short)0, c,
                                                 false, false);
}

__device__ __forceinline__ v8f gemm_core(const u16* __restrict__ A, int lda,
                                         const u16* __restrict__ Bt, int ldb,
                                         int K, int m0, int n0, int lane) {
  v8f acc = {};
  for (int k0 = 0; k0 < K; k0 += 32) {
    v16bf a = load_frag_rm(A, lda, m0, k0, lane);
    v16bf b = load_frag_rm(Bt, ldb, n0, k0, lane);
    acc = wmma_bf16(a, b, acc);
  }
  return acc;
}

// ---------- CDNA5 async global->LDS ----------
__device__ __forceinline__ void async_load_b128(u32 ldsoff, const u32* gaddr) {
  asm volatile("global_load_async_to_lds_b128 %0, %1, off"
               :: "v"(ldsoff), "v"(gaddr) : "memory");
}
__device__ __forceinline__ void wait_async_le4() {
  asm volatile("s_wait_asynccnt 0x4" ::: "memory");
}
__device__ __forceinline__ void wait_async_le0() {
  asm volatile("s_wait_asynccnt 0x0" ::: "memory");
}

// ---------- constants ----------
#define BATCH 128
#define NB 4
#define PP 21
#define NBEH 28
#define CIN 1024
#define CMID 512
#define COUT 256
#define DD 2304
#define M1 4608       // B*NB*9
#define KC 6912       // 3*DD
#define LSTR 132      // LDS row stride (dwords) to dodge bank conflicts

// ---------- prep kernels ----------
__global__ void k_transpose(const float* __restrict__ fmap, u16* __restrict__ fmt) {
  __shared__ float tile[32][33];
  int b = blockIdx.z;
  int yx0 = blockIdx.x * 32;
  int c0 = blockIdx.y * 32;
  int tx = threadIdx.x & 31, ty = threadIdx.x >> 5;
#pragma unroll
  for (int i = 0; i < 4; ++i) {
    int cl = ty + i * 8;
    int yx = yx0 + tx;
    tile[cl][tx] = (yx < 196) ? fmap[((size_t)b * CIN + c0 + cl) * 196 + yx] : 0.f;
  }
  __syncthreads();
#pragma unroll
  for (int i = 0; i < 4; ++i) {
    int yxl = ty + i * 8;
    int yx = yx0 + yxl;
    if (yx < 196)
      fmt[((size_t)b * 196 + yx) * CIN + c0 + tx] = f2bf(tile[tx][yxl]);
  }
}

__global__ void k_cvt(const float* __restrict__ in, u16* __restrict__ out, int n) {
  int i = blockIdx.x * 256 + threadIdx.x;
  if (i < n) out[i] = f2bf(in[i]);
}

__global__ void k_wf2pad(const float* __restrict__ wf2, u16* __restrict__ out) {
  int i = blockIdx.x * 256 + threadIdx.x;
  if (i < 32 * 1024) out[i] = ((i >> 10) < NBEH) ? f2bf(wf2[i]) : (u16)0;
}

// w1d (D, D, 3) fp32 -> wA row-major [o][k*D + c] bf16
__global__ void k_wconv(const float* __restrict__ w, u16* __restrict__ wa) {
  int i = blockIdx.x * 256 + threadIdx.x;
  if (i >= DD * DD) return;
  int o = i / DD, c = i - o * DD;
  const float* s = w + (size_t)i * 3;
#pragma unroll
  for (int k = 0; k < 3; ++k)
    wa[(size_t)o * KC + k * DD + c] = f2bf(s[k]);
}

__global__ void k_zero(u32* __restrict__ p, int n) {
  int i = blockIdx.x * 256 + threadIdx.x;
  if (i < n) p[i] = 0u;
}

// ---------- ROI sampling ----------
__device__ __forceinline__ void axis_setup(float lo, float len, int* i0, int* i1,
                                           float* fr) {
#pragma unroll
  for (int s = 0; s < 6; ++s) {
    float pos = lo + (s + 0.5f) * (1.0f / 6.0f) * len;
    float p = fminf(fmaxf(pos, 0.f), 13.f);
    int a = (int)floorf(p);
    a = a < 0 ? 0 : (a > 13 ? 13 : a);
    i0[s] = a;
    i1[s] = (a < 13) ? a + 1 : 13;
    fr[s] = p - (float)a;
  }
}

__device__ __forceinline__ void roi_accum(const u16* __restrict__ fb, int c,
                                          const int* y0, const int* y1,
                                          const float* ly, const int* x0,
                                          const int* x1, const float* lx,
                                          float* o) {
#pragma unroll
  for (int sy = 0; sy < 6; ++sy) {
    int ry0 = y0[sy] * 14, ry1 = y1[sy] * 14;
    float fy = ly[sy];
#pragma unroll
    for (int sx = 0; sx < 6; ++sx) {
      float fx = lx[sx];
      float v00 = bf2f(fb[(size_t)(ry0 + x0[sx]) * CIN + c]);
      float v01 = bf2f(fb[(size_t)(ry0 + x1[sx]) * CIN + c]);
      float v10 = bf2f(fb[(size_t)(ry1 + x0[sx]) * CIN + c]);
      float v11 = bf2f(fb[(size_t)(ry1 + x1[sx]) * CIN + c]);
      float v = v00 * (1.f - fy) * (1.f - fx) + v01 * (1.f - fy) * fx +
                v10 * fy * (1.f - fx) + v11 * fy * fx;
      o[(sy >> 1) * 3 + (sx >> 1)] += v * 0.25f;
    }
  }
}

__global__ void k_groi(const u16* __restrict__ fmt, float* __restrict__ g) {
  int b = blockIdx.x;
  int y0[6], y1[6], x0[6], x1[6];
  float ly[6], lx[6];
  axis_setup(0.f, 14.f, y0, y1, ly);
  axis_setup(0.f, 14.f, x0, x1, lx);
  const u16* fb = fmt + (size_t)b * 196 * CIN;
  for (int c = threadIdx.x; c < CIN; c += blockDim.x) {
    float o[9] = {0, 0, 0, 0, 0, 0, 0, 0, 0};
    roi_accum(fb, c, y0, y1, ly, x0, x1, lx, o);
#pragma unroll
    for (int ij = 0; ij < 9; ++ij)
      g[((size_t)b * 9 + ij) * CIN + c] = o[ij];
  }
}

__global__ void k_roi(const u16* __restrict__ fmt, const float* __restrict__ g,
                      const float* __restrict__ boxes, u16* __restrict__ X) {
  int bn = blockIdx.x;
  int b = bn >> 2;
  const float* bx = boxes + (size_t)bn * 4;
  float fx1 = fminf(fmaxf(bx[0] * (1.f / 448.f), 0.f), 14.f) * 14.f;
  float fy1 = fminf(fmaxf(bx[1] * (1.f / 448.f), 0.f), 14.f) * 14.f;
  float fx2 = fminf(fmaxf(bx[2] * (1.f / 448.f), 0.f), 14.f) * 14.f;
  float fy2 = fminf(fmaxf(bx[3] * (1.f / 448.f), 0.f), 14.f) * 14.f;
  float rw = fmaxf(fx2 - fx1, 1.f), rh = fmaxf(fy2 - fy1, 1.f);
  int y0[6], y1[6], x0[6], x1[6];
  float ly[6], lx[6];
  axis_setup(fy1, rh, y0, y1, ly);
  axis_setup(fx1, rw, x0, x1, lx);
  const u16* fb = fmt + (size_t)b * 196 * CIN;
  for (int c = threadIdx.x; c < CIN; c += blockDim.x) {
    float o[9] = {0, 0, 0, 0, 0, 0, 0, 0, 0};
    roi_accum(fb, c, y0, y1, ly, x0, x1, lx, o);
#pragma unroll
    for (int ij = 0; ij < 9; ++ij)
      X[((size_t)bn * 9 + ij) * CIN + c] =
          f2bf(o[ij] + g[((size_t)b * 9 + ij) * CIN + c]);
  }
}

// ---------- GEMM kernels (one wave32 per 16x16 tile) ----------
__global__ void k_gemm_bf16(const u16* __restrict__ A, int lda,
                            const u16* __restrict__ Bt, int ldb, int K,
                            u16* __restrict__ C, int ldc) {
  int lane = threadIdx.x;
  int m0 = blockIdx.x * 16, n0 = blockIdx.y * 16;
  v8f acc = gemm_core(A, lda, Bt, ldb, K, m0, n0, lane);
  int hi = (lane & 16) ? 8 : 0;
  int col = n0 + (lane & 15);
#pragma unroll
  for (int j = 0; j < 8; ++j)
    C[(size_t)(m0 + j + hi) * ldc + col] = f2bf(acc[j]);
}

// Y1[4608x512] x wc2b[256x512]^T -> feats fp32 [bn][o*9+ij]
__global__ void k_gemm_feats(const u16* __restrict__ A, const u16* __restrict__ Bt,
                             float* __restrict__ feats) {
  int lane = threadIdx.x;
  int m0 = blockIdx.x * 16, n0 = blockIdx.y * 16;
  v8f acc = gemm_core(A, CMID, Bt, CMID, CMID, m0, n0, lane);
  int hi = (lane & 16) ? 8 : 0;
  int oc = n0 + (lane & 15);
#pragma unroll
  for (int j = 0; j < 8; ++j) {
    int m = m0 + j + hi;
    int bn = m / 9, ij = m - bn * 9;
    feats[(size_t)bn * DD + oc * 9 + ij] = acc[j];
  }
}

// z[512x2304] x wf1b[1024x2304]^T -> relu+bias -> hid bf16
__global__ void k_fc1(const u16* __restrict__ A, const u16* __restrict__ Bt,
                      const float* __restrict__ bias, u16* __restrict__ hid) {
  int lane = threadIdx.x;
  int m0 = blockIdx.x * 16, n0 = blockIdx.y * 16;
  v8f acc = gemm_core(A, DD, Bt, DD, DD, m0, n0, lane);
  int hi = (lane & 16) ? 8 : 0;
  int col = n0 + (lane & 15);
  float bb = bias[col];
#pragma unroll
  for (int j = 0; j < 8; ++j)
    hid[(size_t)(m0 + j + hi) * 1024 + col] = f2bf(fmaxf(acc[j] + bb, 0.f));
}

// hid[512x1024] x wf2b[32x1024]^T -> logits fp32 (valid cols < 28)
__global__ void k_fc2(const u16* __restrict__ A, const u16* __restrict__ Bt,
                      const float* __restrict__ bias, float* __restrict__ out) {
  int lane = threadIdx.x;
  int m0 = blockIdx.x * 16, n0 = blockIdx.y * 16;
  v8f acc = gemm_core(A, 1024, Bt, 1024, 1024, m0, n0, lane);
  int hi = (lane & 16) ? 8 : 0;
  int oc = n0 + (lane & 15);
  if (oc < NBEH) {
    float bb = bias[oc];
#pragma unroll
    for (int j = 0; j < 8; ++j)
      out[(size_t)(m0 + j + hi) * NBEH + oc] = acc[j] + bb;
  }
}

// ---------- scatter feats -> 3 k-shifted paired-channel copies ----------
// seqS[k][p][c2][t] (u32 = bf16 pair {2c,2c+1}) holds value at t_actual = t+k-1
__global__ void k_scatter(const float* __restrict__ feats,
                          const int* __restrict__ pidx, u32* __restrict__ seqS) {
  int bn = blockIdx.x;
  int b = bn >> 2;
  int p = pidx[bn];
  const float* f = feats + (size_t)bn * DD;
  for (int d2 = threadIdx.x; d2 < DD / 2; d2 += blockDim.x) {
    u32 v = pack2(f[2 * d2], f[2 * d2 + 1]);
#pragma unroll
    for (int k = 0; k < 3; ++k) {
      int t = b + 1 - k;
      if (t >= 0 && t < 128)
        seqS[((size_t)(k * PP + p) * (DD / 2) + d2) * 128 + t] = v;
    }
  }
}

// ---------- conv1d: 4 waves/block, async-LDS staged B, double buffered ----------
__device__ __forceinline__ const u32* slice_base(const u32* __restrict__ S, int p,
                                                 int kt) {
  int K0 = kt * 32;
  int k = K0 / DD;
  int c2 = (K0 - k * DD) >> 1;
  return S + ((size_t)(k * PP + p) * (DD / 2) + c2) * 128;
}

__device__ __forceinline__ void stage_slice(const u32* __restrict__ gbase,
                                            u32 sbase, int tid) {
#pragma unroll
  for (int i = 0; i < 4; ++i) {
    int d = i * 512 + tid * 4;                  // 16 rows x 128 dwords
    int row = d >> 7, col = d & 127;
    async_load_b128(sbase + (u32)((row * LSTR + col) * 4), gbase + d);
  }
}

__device__ __forceinline__ void conv_mma_lds(const u16* __restrict__ wA,
                                             const u32* __restrict__ S, int p,
                                             int o0, int tid, int lane,
                                             u32 sa0, u32 sa1,
                                             const u32* sb0, const u32* sb1,
                                             v8f acc[8]) {
  stage_slice(slice_base(S, p, 0), sa0, tid);
  for (int kt = 0; kt < KC / 32; ++kt) {
    if (kt < KC / 32 - 1) {
      stage_slice(slice_base(S, p, kt + 1), ((kt + 1) & 1) ? sa1 : sa0, tid);
      wait_async_le4();
    } else {
      wait_async_le0();
    }
    __syncthreads();
    __builtin_prefetch(wA + (size_t)(o0 + (lane & 15)) * KC + kt * 32 + 64, 0, 1);
    v16bf a = load_frag_rm(wA, KC, o0, kt * 32, lane);
    const u32* sb = (kt & 1) ? sb1 : sb0;
    int c2l = (lane & 16) ? 4 : 0;
    int coll = lane & 15;
#pragma unroll
    for (int tt = 0; tt < 8; ++tt) {
      Frag fb;
#pragma unroll
      for (int v = 0; v < 8; ++v)
        fb.u[v] = sb[(c2l + (v & 3) + ((v >> 2) << 3)) * LSTR + tt * 16 + coll];
      acc[tt] = wmma_bf16(a, fb.v, acc[tt]);
    }
    __syncthreads();
  }
}

__global__ void __launch_bounds__(128) k_conv1(const u16* __restrict__ wA,
                                               const u32* __restrict__ seqS,
                                               const float* __restrict__ bias,
                                               u32* __restrict__ outS) {
  __shared__ u32 sbuf0[16 * LSTR];
  __shared__ u32 sbuf1[16 * LSTR];
  int tid = threadIdx.x, lane = tid & 31, w = tid >> 5;
  int p = blockIdx.x;
  int o0 = blockIdx.y * 64 + w * 16;
  u32 sa0 = (u32)(size_t)&sbuf0[0];
  u32 sa1 = (u32)(size_t)&sbuf1[0];
  v8f acc[8];
#pragma unroll
  for (int t = 0; t < 8; ++t) acc[t] = (v8f){};
  conv_mma_lds(wA, seqS, p, o0, tid, lane, sa0, sa1, sbuf0, sbuf1, acc);

  int hi = (lane & 16) ? 8 : 0;
  int coll = lane & 15;
#pragma unroll
  for (int j = 0; j < 4; ++j) {
    int r0 = o0 + hi + 2 * j;
    float b0 = bias[r0], b1 = bias[r0 + 1];
    int c2 = r0 >> 1;
#pragma unroll
    for (int tt = 0; tt < 8; ++tt) {
      int t = tt * 16 + coll;
      u32 val = pack2(acc[tt][2 * j] + b0, acc[tt][2 * j + 1] + b1);
      outS[((size_t)(1 * PP + p) * (DD / 2) + c2) * 128 + t] = val;        // k=1
      if (t < 127)
        outS[((size_t)(0 * PP + p) * (DD / 2) + c2) * 128 + t + 1] = val;  // k=0
      if (t > 0)
        outS[((size_t)(2 * PP + p) * (DD / 2) + c2) * 128 + t - 1] = val;  // k=2
    }
  }
  if (lane < 8) {  // zero the shifted-out edge columns for this wave's rows
    int c2 = (o0 >> 1) + lane;
    outS[((size_t)(0 * PP + p) * (DD / 2) + c2) * 128 + 0] = 0u;
    outS[((size_t)(2 * PP + p) * (DD / 2) + c2) * 128 + 127] = 0u;
  }
}

__global__ void __launch_bounds__(128) k_conv2(const u16* __restrict__ wA,
                                               const u32* __restrict__ inS,
                                               const float* __restrict__ bias,
                                               float* __restrict__ pf) {
  __shared__ u32 sbuf0[16 * LSTR];
  __shared__ u32 sbuf1[16 * LSTR];
  int tid = threadIdx.x, lane = tid & 31, w = tid >> 5;
  int p = blockIdx.x;
  int o0 = blockIdx.y * 64 + w * 16;
  u32 sa0 = (u32)(size_t)&sbuf0[0];
  u32 sa1 = (u32)(size_t)&sbuf1[0];
  v8f acc[8];
#pragma unroll
  for (int t = 0; t < 8; ++t) acc[t] = (v8f){};
  conv_mma_lds(wA, inS, p, o0, tid, lane, sa0, sa1, sbuf0, sbuf1, acc);

  int hi = (lane & 16) ? 8 : 0;
  int coll = lane & 15;
#pragma unroll
  for (int j = 0; j < 8; ++j) {
    float v = 0.f;
#pragma unroll
    for (int tt = 0; tt < 8; ++tt) v += acc[tt][j];
    v += __shfl_xor(v, 1, 32);
    v += __shfl_xor(v, 2, 32);
    v += __shfl_xor(v, 4, 32);
    v += __shfl_xor(v, 8, 32);
    if (coll == 0) {
      int r = o0 + hi + j;
      pf[(size_t)p * DD + r] = v * (1.f / 128.f) + bias[r];
    }
  }
}

__global__ void k_z(const float* __restrict__ feats, const float* __restrict__ pf,
                    const int* __restrict__ pidx, u16* __restrict__ z) {
  int bn = blockIdx.x;
  int p = pidx[bn];
  const float* f = feats + (size_t)bn * DD;
  const float* q = pf + (size_t)p * DD;
  for (int d = threadIdx.x; d < DD; d += blockDim.x)
    z[(size_t)bn * DD + d] = f2bf(f[d] + q[d]);
}

// ---------- launch ----------
extern "C" void kernel_launch(void* const* d_in, const int* in_sizes, int n_in,
                              void* d_out, int out_size, void* d_ws, size_t ws_size,
                              hipStream_t stream) {
  (void)in_sizes; (void)n_in; (void)out_size; (void)ws_size;
  const float* fmap  = (const float*)d_in[0];
  const float* boxes = (const float*)d_in[1];
  const int*   pidx  = (const int*)d_in[2];
  const float* w_c1  = (const float*)d_in[3];
  const float* w_c2  = (const float*)d_in[4];
  const float* w1d1  = (const float*)d_in[5];
  const float* b1d1  = (const float*)d_in[6];
  const float* w1d2  = (const float*)d_in[7];
  const float* b1d2  = (const float*)d_in[8];
  const float* wf1   = (const float*)d_in[9];
  const float* bf1   = (const float*)d_in[10];
  const float* wf2   = (const float*)d_in[11];
  const float* bf2   = (const float*)d_in[12];
  float* logits = (float*)d_out;

  char* w = (char*)d_ws;
  size_t off = 0;
  auto alloc = [&](size_t n) -> void* {
    void* p = w + off;
    off += (n + 255) & ~(size_t)255;
    return p;
  };
  u16*  fmt   = (u16*)alloc((size_t)BATCH * 196 * CIN * 2);
  float* g    = (float*)alloc((size_t)BATCH * 9 * CIN * 4);
  u16*  X     = (u16*)alloc((size_t)M1 * CIN * 2);
  u16*  wc1b  = (u16*)alloc((size_t)CMID * CIN * 2);
  u16*  wc2b  = (u16*)alloc((size_t)COUT * CMID * 2);
  u16*  Y1    = (u16*)alloc((size_t)M1 * CMID * 2);
  float* feats = (float*)alloc((size_t)BATCH * NB * DD * 4);
  u16*  wA1   = (u16*)alloc((size_t)DD * KC * 2);
  u16*  wA2   = (u16*)alloc((size_t)DD * KC * 2);
  u32*  seqS  = (u32*)alloc((size_t)3 * PP * (DD / 2) * 128 * 4);
  u32*  h1S   = (u32*)alloc((size_t)3 * PP * (DD / 2) * 128 * 4);
  float* pf   = (float*)alloc((size_t)PP * DD * 4);
  u16*  z     = (u16*)alloc((size_t)BATCH * NB * DD * 2);
  u16*  wf1b  = (u16*)alloc((size_t)1024 * DD * 2);
  u16*  wf2b  = (u16*)alloc((size_t)32 * 1024 * 2);
  u16*  hid   = (u16*)alloc((size_t)BATCH * NB * 1024 * 2);

  // weight/layout prep
  k_transpose<<<dim3(7, 32, BATCH), 256, 0, stream>>>(fmap, fmt);
  k_cvt<<<(CMID * CIN + 255) / 256, 256, 0, stream>>>(w_c1, wc1b, CMID * CIN);
  k_cvt<<<(COUT * CMID + 255) / 256, 256, 0, stream>>>(w_c2, wc2b, COUT * CMID);
  k_cvt<<<(1024 * DD + 255) / 256, 256, 0, stream>>>(wf1, wf1b, 1024 * DD);
  k_wf2pad<<<(32 * 1024 + 255) / 256, 256, 0, stream>>>(wf2, wf2b);
  k_wconv<<<(DD * DD + 255) / 256, 256, 0, stream>>>(w1d1, wA1);
  k_wconv<<<(DD * DD + 255) / 256, 256, 0, stream>>>(w1d2, wA2);
  int seqN = 3 * PP * (DD / 2) * 128;
  k_zero<<<(seqN + 255) / 256, 256, 0, stream>>>(seqS, seqN);

  // ROI feature extraction
  k_groi<<<BATCH, 256, 0, stream>>>(fmt, g);
  k_roi<<<BATCH * NB, 256, 0, stream>>>(fmt, g, boxes, X);

  // channel mixing GEMMs (WMMA bf16)
  k_gemm_bf16<<<dim3(M1 / 16, CMID / 16), 32, 0, stream>>>(X, CIN, wc1b, CIN, CIN,
                                                           Y1, CMID);
  k_gemm_feats<<<dim3(M1 / 16, COUT / 16), 32, 0, stream>>>(Y1, wc2b, feats);

  // temporal convs (WMMA bf16, async-LDS staged B operand)
  k_scatter<<<BATCH * NB, 256, 0, stream>>>(feats, pidx, seqS);
  k_conv1<<<dim3(PP, DD / 64), 128, 0, stream>>>(wA1, seqS, b1d1, h1S);
  k_conv2<<<dim3(PP, DD / 64), 128, 0, stream>>>(wA2, h1S, b1d2, pf);

  // head
  k_z<<<BATCH * NB, 256, 0, stream>>>(feats, pf, pidx, z);
  k_fc1<<<dim3((BATCH * NB) / 16, 1024 / 16), 32, 0, stream>>>(z, wf1b, bf1, hid);
  k_fc2<<<dim3((BATCH * NB) / 16, 2), 32, 0, stream>>>(hid, wf2b, bf2, logits);
}
